// MultiHeadAttention_36713380446695
// MI455X (gfx1250) — matrix operand order repaired and checked by reading, not compiled, not run on hardware
//
#include <hip/hip_runtime.h>
#include <math.h>

typedef __bf16 bf16;
typedef __attribute__((ext_vector_type(8)))  __bf16 v8bf;
typedef __attribute__((ext_vector_type(16))) __bf16 v16bf;
typedef __attribute__((ext_vector_type(8)))  float  v8f;

#define BF_WMMA(A, B, C) \
  __builtin_amdgcn_wmma_f32_16x16x32_bf16(false, (A), false, (B), (short)0, (C), false, false)

// s_wait_asynccnt with an immediate (SOPP); N must be a string literal
#define S_WAIT_ASYNCCNT(N) asm volatile("s_wait_asynccnt " N ::: "memory")

constexpr int Bn = 4, Hn = 16, Dh = 64, Ls = 1024, Es = 1024, Mrel = 1024;
constexpr int PEVW = 2112;          // padded row width (elements) for pev^T table
constexpr int PAD  = 32;            // zero margin in p_lds rows
constexpr int PW   = 96;            // p_lds row width: 32 margin + 32 data + 32 margin

static __device__ inline v16bf concat8(v8bf lo, v8bf hi) {
  return __builtin_shufflevector(lo, hi, 0,1,2,3,4,5,6,7,8,9,10,11,12,13,14,15);
}

// Async 16-byte copy global -> LDS (per lane). VDST VGPR carries the LDS byte
// offset, VADDR the 64-bit global address (VGLOBAL async encoding, ASYNCcnt).
static __device__ inline void async_copy16(const bf16* __restrict__ gsrc, bf16* ldst) {
  unsigned int loff =
      (unsigned int)(unsigned long long)(__attribute__((address_space(3))) bf16*)ldst;
  unsigned long long gaddr = (unsigned long long)(unsigned long)gsrc;
  asm volatile("global_load_async_to_lds_b128 %0, %1, off"
               :: "v"(loff), "v"(gaddr) : "memory");
}

// A fragment (16x32 bf16). Lane L: row = L&15, half = L>>4.
// Elements 0..7  = K = 8*half + {0..7};  elements 8..15 = K = 16 + 8*half + {0..7}.
static __device__ inline v16bf load_a_frag(const bf16* __restrict__ base, int lda,
                                           int row0, int k0, int lane) {
  const int r = row0 + (lane & 15);
  const int half = lane >> 4;
  const bf16* p = base + (size_t)r * lda + k0 + 8 * half;
  v8bf lo = *(const v8bf*)p;
  v8bf hi = *(const v8bf*)(p + 16);
  return concat8(lo, hi);
}

// Same fragment but from an LDS-staged [128][32] tile.
static __device__ inline v16bf load_a_frag_lds(const bf16* base, int row0, int lane) {
  const int r = row0 + (lane & 15);
  const int half = lane >> 4;
  const bf16* p = base + r * 32 + 8 * half;
  v8bf lo = *(const v8bf*)p;
  v8bf hi = *(const v8bf*)(p + 16);
  return concat8(lo, hi);
}

// B fragment (32x16 bf16) from Bt stored [N x K] row-major.
// Lane L: col = L&15, half = L>>4; elements 0..7 = K = 8*half + {0..7}; 8..15 = K = 16 + 8*half + {0..7}.
static __device__ inline v16bf load_b_frag(const bf16* __restrict__ bt, int ldb,
                                           int col0, int k0, int lane) {
  const int n = col0 + (lane & 15);
  const int half = lane >> 4;
  const bf16* p = bt + (size_t)n * ldb + k0 + 8 * half;
  v8bf lo = *(const v8bf*)p;
  v8bf hi = *(const v8bf*)(p + 16);
  return concat8(lo, hi);
}

// ---------------------------------------------------------------- casts
__global__ void cast_f32_bf16(const float* __restrict__ src, bf16* __restrict__ dst, int n) {
  int i = blockIdx.x * blockDim.x + threadIdx.x;
  int stride = gridDim.x * blockDim.x;
  for (; i < n; i += stride) dst[i] = (bf16)src[i];
}

__global__ void transpose_pev(const float* __restrict__ pev, bf16* __restrict__ pevT) {
  int j = blockIdx.x * blockDim.x + threadIdx.x;
  if (j >= PEVW) return;
  for (int d = 0; d < Dh; ++d)
    pevT[(size_t)d * PEVW + j] = (j < 2 * Mrel + 1) ? (bf16)pev[(size_t)j * Dh + d] : (bf16)0.f;
}

// ---------------------------------------------------------------- fused QKV GEMM
// C[m,e] = sum_c xb[m,c] * W[e,c]; A staged in LDS via async copies (double
// buffered on ASYNCcnt), Bt = W streamed from L2.
// z=0 -> Q (B,H,L,D), z=1 -> K (B,H,L,D), z=2 -> V stored transposed (B,H,D,L).
__global__ __launch_bounds__(256) void gemm_qkv_kernel(
    const bf16* __restrict__ xb, const bf16* __restrict__ wq,
    const bf16* __restrict__ wk, const bf16* __restrict__ wv,
    bf16* __restrict__ Qo, bf16* __restrict__ Ko, bf16* __restrict__ VTo) {
  __shared__ __align__(16) bf16 At[2][128][32];
  const int tid  = threadIdx.x;
  const int lane = tid & 31;
  const int wave = tid >> 5;
  const int mblk = blockIdx.x * 128;
  const int m0w  = (wave >> 1) * 32;                 // wave's rows within block
  const int n0   = blockIdx.y * 64 + (wave & 1) * 32;
  const int z    = blockIdx.z;
  const bf16* W  = (z == 0) ? wq : (z == 1) ? wk : wv;

  // 512 16B chunks per 128x32 tile; 2 per thread
  const int r0 = tid >> 2,          p0 = (tid & 3) * 8;
  const int r1 = (tid + 256) >> 2,  p1 = (tid & 3) * 8;   // (tid+256)&3 == tid&3

  async_copy16(xb + (size_t)(mblk + r0) * Es + p0, &At[0][r0][p0]);
  async_copy16(xb + (size_t)(mblk + r1) * Es + p1, &At[0][r1][p1]);

  v8f c00{}, c01{}, c10{}, c11{};
  for (int kk = 0; kk < Es; kk += 32) {
    const int buf = (kk >> 5) & 1;
    if (kk + 32 < Es) {
      async_copy16(xb + (size_t)(mblk + r0) * Es + kk + 32 + p0, &At[buf ^ 1][r0][p0]);
      async_copy16(xb + (size_t)(mblk + r1) * Es + kk + 32 + p1, &At[buf ^ 1][r1][p1]);
      S_WAIT_ASYNCCNT("2");       // current buffer's 2 copies complete (in-order)
    } else {
      S_WAIT_ASYNCCNT("0");
    }
    __syncthreads();
    __builtin_prefetch(W + (size_t)(n0 + (lane & 15)) * Es + kk + 64, 0, 1);
    v16bf a0 = load_a_frag_lds(&At[buf][0][0], m0w,      lane);
    v16bf a1 = load_a_frag_lds(&At[buf][0][0], m0w + 16, lane);
    v16bf b0 = load_b_frag(W, Es, n0,      kk, lane);
    v16bf b1 = load_b_frag(W, Es, n0 + 16, kk, lane);
    c00 = BF_WMMA(a0, b0, c00);
    c01 = BF_WMMA(a0, b1, c01);
    c10 = BF_WMMA(a1, b0, c10);
    c11 = BF_WMMA(a1, b1, c11);
  }
  const int colL = lane & 15, half = lane >> 4;
  for (int f = 0; f < 4; ++f) {
    const v8f& c = (f == 0) ? c00 : (f == 1) ? c01 : (f == 2) ? c10 : c11;
    const int mb = mblk + m0w + (f >> 1) * 16;
    const int nb = n0 + (f & 1) * 16;
    for (int i = 0; i < 8; ++i) {
      const int m = mb + i + 8 * half;
      const int e = nb + colL;
      const int b = m >> 10, l = m & 1023;
      const int h = e >> 6,  d = e & 63;
      const bf16 v = (bf16)c[i];
      if (z == 0)      Qo [(((size_t)(b * Hn + h)) * Ls + l) * Dh + d] = v;
      else if (z == 1) Ko [(((size_t)(b * Hn + h)) * Ls + l) * Dh + d] = v;
      else             VTo[(((size_t)(b * Hn + h)) * Dh + d) * Ls + l] = v;
    }
  }
}

// ---------------------------------------------------------------- output projection GEMM
__global__ __launch_bounds__(256) void gemm_out_kernel(
    const bf16* __restrict__ ab, const bf16* __restrict__ wo, float* __restrict__ out) {
  __shared__ __align__(16) bf16 At[2][128][32];
  const int tid  = threadIdx.x;
  const int lane = tid & 31;
  const int wave = tid >> 5;
  const int mblk = blockIdx.x * 128;
  const int m0w  = (wave >> 1) * 32;
  const int n0   = blockIdx.y * 64 + (wave & 1) * 32;

  const int r0 = tid >> 2,          p0 = (tid & 3) * 8;
  const int r1 = (tid + 256) >> 2,  p1 = (tid & 3) * 8;

  async_copy16(ab + (size_t)(mblk + r0) * Es + p0, &At[0][r0][p0]);
  async_copy16(ab + (size_t)(mblk + r1) * Es + p1, &At[0][r1][p1]);

  v8f c00{}, c01{}, c10{}, c11{};
  for (int kk = 0; kk < Es; kk += 32) {
    const int buf = (kk >> 5) & 1;
    if (kk + 32 < Es) {
      async_copy16(ab + (size_t)(mblk + r0) * Es + kk + 32 + p0, &At[buf ^ 1][r0][p0]);
      async_copy16(ab + (size_t)(mblk + r1) * Es + kk + 32 + p1, &At[buf ^ 1][r1][p1]);
      S_WAIT_ASYNCCNT("2");
    } else {
      S_WAIT_ASYNCCNT("0");
    }
    __syncthreads();
    __builtin_prefetch(wo + (size_t)(n0 + (lane & 15)) * Es + kk + 64, 0, 1);
    v16bf a0 = load_a_frag_lds(&At[buf][0][0], m0w,      lane);
    v16bf a1 = load_a_frag_lds(&At[buf][0][0], m0w + 16, lane);
    v16bf b0 = load_b_frag(wo, Es, n0,      kk, lane);
    v16bf b1 = load_b_frag(wo, Es, n0 + 16, kk, lane);
    c00 = BF_WMMA(a0, b0, c00);
    c01 = BF_WMMA(a0, b1, c01);
    c10 = BF_WMMA(a1, b0, c10);
    c11 = BF_WMMA(a1, b1, c11);
  }
  const int colL = lane & 15, half = lane >> 4;
  for (int f = 0; f < 4; ++f) {
    const v8f& c = (f == 0) ? c00 : (f == 1) ? c01 : (f == 2) ? c10 : c11;
    const int mb = mblk + m0w + (f >> 1) * 16;
    const int nb = n0 + (f & 1) * 16;
    for (int i = 0; i < 8; ++i)
      out[(size_t)(mb + i + 8 * half) * Es + nb + colL] = c[i];
  }
}

// ---------------------------------------------------------------- flash attention + RPE
// One wave per 16-row q tile; 4 waves per block (independent, no block barriers).
__global__ __launch_bounds__(128) void attn_kernel(
    const bf16* __restrict__ Q, const bf16* __restrict__ K,
    const bf16* __restrict__ VT, const bf16* __restrict__ pekb,
    const bf16* __restrict__ pevT, bf16* __restrict__ ctx) {
  __shared__ float qp_lds[4][16][32];
  __shared__ __align__(16) bf16 p_lds[4][16][PW];

  const int lane = threadIdx.x & 31;
  const int wave = threadIdx.x >> 5;
  const int tile = blockIdx.x * 4 + wave;           // B*H*(L/16) = 4096 tiles
  const int b  = tile >> 10;
  const int h  = (tile >> 6) & 15;
  const int q0 = (tile & 63) << 4;
  const int colL = lane & 15, half = lane >> 4;

  // zero this wave's p_lds (data + margins)
  for (int t = lane; t < 16 * PW; t += 32) (&p_lds[wave][0][0])[t] = (bf16)0.f;

  const bf16* Qb = Q  + ((size_t)(b * Hn + h)) * Ls * Dh;
  const bf16* Kb = K  + ((size_t)(b * Hn + h)) * Ls * Dh;
  const bf16* Vb = VT + ((size_t)(b * Hn + h)) * Dh * Ls;

  const v16bf qa0 = load_a_frag(Qb, Dh, q0, 0,  lane);
  const v16bf qa1 = load_a_frag(Qb, Dh, q0, 32, lane);

  v8f o0{}, o1{}, o2{}, o3{};
  float mrow[8], lrow[8];
  for (int i = 0; i < 8; ++i) { mrow[i] = -1e30f; lrow[i] = 0.f; }

  for (int k0 = 0; k0 < q0 + 16; k0 += 32) {
    for (int sub = 0; sub < 2; ++sub) {
      const int k0s = k0 + 16 * sub;
      if (k0s >= q0 + 16) {     // fully beyond causal frontier: zero the P slot
        for (int i = 0; i < 8; ++i)
          p_lds[wave][i + 8 * half][PAD + 16 + colL] = (bf16)0.f;
        continue;
      }
      // content score S = Q K^T (contract over d = 64)
      v8f s{};
      s = BF_WMMA(qa0, load_b_frag(Kb, Dh, k0s, 0,  lane), s);
      s = BF_WMMA(qa1, load_b_frag(Kb, Dh, k0s, 32, lane), s);

      // rel-key window Qp[q, j) for j in [jb, jb+32), jb = k0s - q0 - 15 + M (>= 1 here)
      const int jb = k0s - q0 - 15 + Mrel;
      v8f qpA{}, qpB{};
      qpA = BF_WMMA(qa0, load_b_frag(pekb, Dh, jb,      0,  lane), qpA);
      qpA = BF_WMMA(qa1, load_b_frag(pekb, Dh, jb,      32, lane), qpA);
      qpB = BF_WMMA(qa0, load_b_frag(pekb, Dh, jb + 16, 0,  lane), qpB);
      qpB = BF_WMMA(qa1, load_b_frag(pekb, Dh, jb + 16, 32, lane), qpB);
      for (int i = 0; i < 8; ++i) {
        qp_lds[wave][i + 8 * half][colL]      = qpA[i];
        qp_lds[wave][i + 8 * half][16 + colL] = qpB[i];
      }

      // gather bias (diagonal shift), mask, scale; online softmax update
      float sc[8];
      for (int i = 0; i < 8; ++i) {
        const int r = i + 8 * half;
        const int q = q0 + r;
        const int k = k0s + colL;
        const int jslot = colL - r + 15;                 // in [0,30]
        const float v = (s[i] + qp_lds[wave][r][jslot]) * 0.125f; // / sqrt(64)
        sc[i] = (k <= q) ? v : -1e30f;
      }
      for (int i = 0; i < 8; ++i) {
        float rm = sc[i];
        rm = fmaxf(rm, __shfl_xor(rm, 1, 32));
        rm = fmaxf(rm, __shfl_xor(rm, 2, 32));
        rm = fmaxf(rm, __shfl_xor(rm, 4, 32));
        rm = fmaxf(rm, __shfl_xor(rm, 8, 32));
        const float mnew  = fmaxf(mrow[i], rm);
        const float alpha = __expf(mrow[i] - mnew);
        const float p     = __expf(sc[i] - mnew);
        float rs = p;
        rs += __shfl_xor(rs, 1, 32);
        rs += __shfl_xor(rs, 2, 32);
        rs += __shfl_xor(rs, 4, 32);
        rs += __shfl_xor(rs, 8, 32);
        lrow[i] = lrow[i] * alpha + rs;
        mrow[i] = mnew;
        o0[i] *= alpha; o1[i] *= alpha; o2[i] *= alpha; o3[i] *= alpha;
        p_lds[wave][i + 8 * half][PAD + (k0s - k0) + colL] = (bf16)p;
      }
    }

    // P @ V over this 32-wide k block (A from LDS, B = V^T rows)
    {
      const bf16* prow = &p_lds[wave][colL][0];
      v8bf lo = *(const v8bf*)(prow + PAD + 8 * half);
      v8bf hi = *(const v8bf*)(prow + PAD + 16 + 8 * half);
      const v16bf pa = concat8(lo, hi);
      o0 = BF_WMMA(pa, load_b_frag(Vb, Ls, 0,  k0, lane), o0);
      o1 = BF_WMMA(pa, load_b_frag(Vb, Ls, 16, k0, lane), o1);
      o2 = BF_WMMA(pa, load_b_frag(Vb, Ls, 32, k0, lane), o2);
      o3 = BF_WMMA(pa, load_b_frag(Vb, Ls, 48, k0, lane), o3);
    }
    // rel-value: two aligned 32-wide j windows cover the 47-wide diagonal band;
    // A built by reading P shifted (margin zeros kill out-of-block k).
    {
      const int jbase = (k0 - q0 - 15 + Mrel) & ~15;    // 16-aligned, >= 0
      for (int w = 0; w < 2; ++w) {
        const int jbw = jbase + 32 * w;
        const int delta = jbw - (k0 - q0 + Mrel);       // shift: col = r + delta + jslot
        const int r = colL;
        const bf16* prow = &p_lds[wave][r][0];
        const int base0 = PAD + r + delta;
        v16bf ps;
        for (int t = 0; t < 8; ++t) ps[t]     = prow[base0 + 8 * half + t];
        for (int t = 0; t < 8; ++t) ps[8 + t] = prow[base0 + 16 + 8 * half + t];
        o0 = BF_WMMA(ps, load_b_frag(pevT, PEVW, 0,  jbw, lane), o0);
        o1 = BF_WMMA(ps, load_b_frag(pevT, PEVW, 16, jbw, lane), o1);
        o2 = BF_WMMA(ps, load_b_frag(pevT, PEVW, 32, jbw, lane), o2);
        o3 = BF_WMMA(ps, load_b_frag(pevT, PEVW, 48, jbw, lane), o3);
      }
    }
  }

  // epilogue: normalize and scatter into (B, L, E) bf16 ctx
  for (int i = 0; i < 8; ++i) {
    const int q = q0 + i + 8 * half;
    const float inv = 1.f / lrow[i];
    bf16* dst = ctx + ((size_t)(b * Ls + q)) * Es + h * Dh;
    dst[0  + colL] = (bf16)(o0[i] * inv);
    dst[16 + colL] = (bf16)(o1[i] * inv);
    dst[32 + colL] = (bf16)(o2[i] * inv);
    dst[48 + colL] = (bf16)(o3[i] * inv);
  }
}

// ---------------------------------------------------------------- host launcher
extern "C" void kernel_launch(void* const* d_in, const int* in_sizes, int n_in,
                              void* d_out, int out_size, void* d_ws, size_t ws_size,
                              hipStream_t stream) {
  const float* x   = (const float*)d_in[0];
  const float* Wq  = (const float*)d_in[1];
  const float* Wk  = (const float*)d_in[2];
  const float* Wv  = (const float*)d_in[3];
  const float* Wo  = (const float*)d_in[4];
  const float* pek = (const float*)d_in[5];
  const float* pev = (const float*)d_in[6];
  float* out = (float*)d_out;
  (void)in_sizes; (void)n_in; (void)out_size; (void)ws_size;

  char* ws = (char*)d_ws;
  size_t off = 0;
  auto carve = [&](size_t bytes) -> void* {
    void* p = ws + off;
    off += (bytes + 255) & ~(size_t)255;
    return p;
  };
  bf16* xb    = (bf16*)carve((size_t)Bn * Ls * Es * 2);
  bf16* wqb   = (bf16*)carve((size_t)Es * Es * 2);
  bf16* wkb   = (bf16*)carve((size_t)Es * Es * 2);
  bf16* wvb   = (bf16*)carve((size_t)Es * Es * 2);
  bf16* wob   = (bf16*)carve((size_t)Es * Es * 2);
  bf16* pekb  = (bf16*)carve((size_t)(2 * Mrel + 1) * Dh * 2);
  bf16* pevTb = (bf16*)carve((size_t)Dh * PEVW * 2);
  bf16* Qb    = (bf16*)carve((size_t)Bn * Hn * Ls * Dh * 2);
  bf16* Kb    = (bf16*)carve((size_t)Bn * Hn * Ls * Dh * 2);
  bf16* VTb   = (bf16*)carve((size_t)Bn * Hn * Dh * Ls * 2);
  bf16* ctxb  = (bf16*)carve((size_t)Bn * Ls * Es * 2);

  cast_f32_bf16<<<2048, 256, 0, stream>>>(x,   xb,  Bn * Ls * Es);
  cast_f32_bf16<<<512,  256, 0, stream>>>(Wq,  wqb, Es * Es);
  cast_f32_bf16<<<512,  256, 0, stream>>>(Wk,  wkb, Es * Es);
  cast_f32_bf16<<<512,  256, 0, stream>>>(Wv,  wvb, Es * Es);
  cast_f32_bf16<<<512,  256, 0, stream>>>(Wo,  wob, Es * Es);
  cast_f32_bf16<<<256,  256, 0, stream>>>(pek, pekb, (2 * Mrel + 1) * Dh);
  transpose_pev<<<(PEVW + 255) / 256, 256, 0, stream>>>(pev, pevTb);

  gemm_qkv_kernel<<<dim3(32, 16, 3), 256, 0, stream>>>(xb, wqb, wkb, wvb, Qb, Kb, VTb);
  attn_kernel<<<dim3(1024), 128, 0, stream>>>(Qb, Kb, VTb, pekb, pevTb, ctxb);
  gemm_out_kernel<<<dim3(32, 16, 1), 256, 0, stream>>>(ctxb, wob, out);
}
